// LuongAttention_17815524344562
// MI455X (gfx1250) — matrix operand order repaired and checked by reading, not compiled, not run on hardware
//
#include <hip/hip_runtime.h>
#include <hip/hip_bf16.h>

// Problem constants (match reference): B=8, TQ=2048, TK=2048, D=1024
#define B_   8
#define TQ_  2048
#define TK_  2048
#define D_   1024

typedef __attribute__((ext_vector_type(16))) __bf16 v16bf;
typedef __attribute__((ext_vector_type(8)))  float  v8f;

union BF16x16 {
  v16bf v;
  uint4 q[2];
  unsigned int u[8];
  unsigned short h[16];
};

// ---- f32 pair -> packed bf16x2 (cheap round-half-up + byte-perm merge) ----
#if __has_builtin(__builtin_amdgcn_cvt_pk_bf16_f32)
typedef __attribute__((ext_vector_type(2))) __bf16 v2bf;
__device__ __forceinline__ unsigned int pack2(float lo, float hi) {
  union { v2bf v; unsigned int u; } c;
  c.v = __builtin_amdgcn_cvt_pk_bf16_f32(lo, hi);
  return c.u;
}
#else
__device__ __forceinline__ unsigned int pack2(float lo, float hi) {
  union { float f; unsigned int u; } a, b;
  a.f = lo; b.f = hi;
  const unsigned int au = a.u + 0x8000u;   // round-half-up into bf16
  const unsigned int bu = b.u + 0x8000u;
#if __has_builtin(__builtin_amdgcn_perm)
  // result = {bu.b3, bu.b2, au.b3, au.b2}  -> one v_perm_b32
  return __builtin_amdgcn_perm(bu, au, 0x07060302u);
#else
  return (bu & 0xFFFF0000u) | (au >> 16);
#endif
}
#endif

__device__ __forceinline__ void wait_dscnt0() {
#if __has_builtin(__builtin_amdgcn_s_wait_dscnt)
  __builtin_amdgcn_s_wait_dscnt(0);
#else
  asm volatile("s_wait_dscnt 0x0" ::: "memory");
#endif
}
__device__ __forceinline__ void wait_asynccnt0() {
#if __has_builtin(__builtin_amdgcn_s_wait_asynccnt)
  __builtin_amdgcn_s_wait_asynccnt(0);
#else
  asm volatile("s_wait_asynccnt 0x0" ::: "memory");
#endif
}

// ---- LDS layout (dynamic shared, region-aliased) --------------------------
// Region A [0, 131328): sS  = f32 [16][2052]   (scores / unnorm P)
//                 then: sVt = u32 [1024][20]   (transposed bf16 V chunk)
// Region B [131328, 197120): sQ = bf16 [16][1032]  then  sP = bf16 [16][2056]
// Region C [197120, 198272): sM[16] f32, sL[16] f32, sRed[256] f32
#define SS_STRIDE    2052
#define SQ_STRIDE    1032
#define SP_STRIDE    2056
#define SVT_STRIDE_U 20
#define REGION_B_OFF 131328
#define REGION_C_OFF 197120
#define SMEM_BYTES   198272

__global__ void __launch_bounds__(256, 1)
LuongAttention_17815524344562_kernel(const float* __restrict__ Qm,
                                     const float* __restrict__ Em,
                                     float* __restrict__ ctx,
                                     float* __restrict__ aln) {
  extern __shared__ unsigned char smem[];
  float*          sS  = (float*)smem;
  unsigned int*   sVt = (unsigned int*)smem;
  unsigned short* sQ  = (unsigned short*)(smem + REGION_B_OFF);
  unsigned short* sP  = (unsigned short*)(smem + REGION_B_OFF);
  float* sM   = (float*)(smem + REGION_C_OFF);
  float* sL   = sM + 16;
  float* sRed = sL + 16;

  const int t    = threadIdx.x;
  const int wave = t >> 5;        // 0..7
  const int lane = t & 31;
  const int hw   = lane >> 4;     // half-wave select (WMMA lane-group)
  const int lq   = lane & 15;
  const int q0   = blockIdx.x * 16;
  const int b    = blockIdx.y;

  const float* Qb = Qm + ((size_t)b * TQ_ + q0) * D_;
  const float* Eb = Em + (size_t)b * TK_ * D_;

  // ---- Stage Q block into LDS as bf16 -------------------------------------
  {
    const int m  = t >> 4;
    const int d0 = (t & 15) * 64;
    const float* src = Qb + (size_t)m * D_ + d0;
    unsigned int* dst = (unsigned int*)(sQ + (size_t)m * SQ_STRIDE) + (d0 >> 1);
    #pragma unroll
    for (int i = 0; i < 64; i += 4) {
      float4 x = *(const float4*)(src + i);
      dst[(i >> 1) + 0] = pack2(x.x, x.y);
      dst[(i >> 1) + 1] = pack2(x.z, x.w);
    }
  }
  __syncthreads();

  // ---- Phase A: S = Q @ K^T -----------------------------------------------
  // Each wave owns 8 *pairs* of adjacent 16-wide k-tiles: one A (Q) tile load
  // feeds two independent WMMAs (acc0/acc1), halving LDS A-traffic per WMMA.
  {
    const uint4* qa = (const uint4*)(sQ + (size_t)lq * SQ_STRIDE);
    for (int j = 0; j < 8; ++j) {
      const int kbase = 32 * wave + 256 * j;       // pair covers [kbase, kbase+32)
      const float* kptr0 = Eb + (size_t)(kbase + lq) * D_ + hw * 16;
      const float* kptr1 = kptr0 + (size_t)16 * D_;
      __builtin_prefetch(kptr0 + (size_t)256 * D_, 0, 0);  // next pair's K rows
      v8f acc0 = {}, acc1 = {};
      for (int d0 = 0; d0 < D_; d0 += 32) {
        BF16x16 a, b0, b1;
        // A (16x32 bf16): lane = row lq, chunks at d0+16*hw and d0+16*hw+8
        a.q[0] = qa[(d0 + 16 * hw) >> 3];
        a.q[1] = qa[(d0 + 16 * hw + 8) >> 3];
        // B tiles (32x16 bf16): lane = k-col, 16 consecutive d from K rows
        float4 f0 = *(const float4*)(kptr0 + d0 + 0);
        float4 f1 = *(const float4*)(kptr0 + d0 + 4);
        float4 f2 = *(const float4*)(kptr0 + d0 + 8);
        float4 f3 = *(const float4*)(kptr0 + d0 + 12);
        float4 g0 = *(const float4*)(kptr1 + d0 + 0);
        float4 g1 = *(const float4*)(kptr1 + d0 + 4);
        float4 g2 = *(const float4*)(kptr1 + d0 + 8);
        float4 g3 = *(const float4*)(kptr1 + d0 + 12);
        b0.u[0] = pack2(f0.x, f0.y); b0.u[1] = pack2(f0.z, f0.w);
        b0.u[2] = pack2(f1.x, f1.y); b0.u[3] = pack2(f1.z, f1.w);
        b0.u[4] = pack2(f2.x, f2.y); b0.u[5] = pack2(f2.z, f2.w);
        b0.u[6] = pack2(f3.x, f3.y); b0.u[7] = pack2(f3.z, f3.w);
        b1.u[0] = pack2(g0.x, g0.y); b1.u[1] = pack2(g0.z, g0.w);
        b1.u[2] = pack2(g1.x, g1.y); b1.u[3] = pack2(g1.z, g1.w);
        b1.u[4] = pack2(g2.x, g2.y); b1.u[5] = pack2(g2.z, g2.w);
        b1.u[6] = pack2(g3.x, g3.y); b1.u[7] = pack2(g3.z, g3.w);
        acc0 = __builtin_amdgcn_wmma_f32_16x16x32_bf16(false, a.v, false, b0.v,
                                                       (short)0, acc0, false, false);
        acc1 = __builtin_amdgcn_wmma_f32_16x16x32_bf16(false, a.v, false, b1.v,
                                                       (short)0, acc1, false, false);
      }
      #pragma unroll
      for (int r = 0; r < 8; ++r) {
        sS[(size_t)(r + 8 * hw) * SS_STRIDE + kbase + lq]      = acc0[r];
        sS[(size_t)(r + 8 * hw) * SS_STRIDE + kbase + 16 + lq] = acc1[r];
      }
    }
  }
  __syncthreads();

  // ---- Softmax over full rows held in LDS ---------------------------------
  const int m   = t >> 4;
  const int seg = (t & 15) * 128;
  float* srow = sS + (size_t)m * SS_STRIDE + seg;

  { // pass 1: row max
    float mx = -3.402823466e38f;
    for (int i = 0; i < 128; i += 4) {
      float4 x = *(const float4*)(srow + i);
      mx = fmaxf(mx, fmaxf(fmaxf(x.x, x.y), fmaxf(x.z, x.w)));
    }
    sRed[t] = mx;
  }
  __syncthreads();
  if (t < 16) {
    float mx = sRed[t * 16];
    for (int j = 1; j < 16; ++j) mx = fmaxf(mx, sRed[t * 16 + j]);
    sM[t] = mx;
  }
  __syncthreads();

  { // pass 2: exp in place, bf16 copy of P, row sum
    const float mx = sM[m];
    unsigned int* prow = (unsigned int*)(sP + (size_t)m * SP_STRIDE) + (seg >> 1);
    float sum = 0.f;
    for (int i = 0; i < 128; i += 4) {
      float4 x = *(const float4*)(srow + i);
      x.x = __expf(x.x - mx); x.y = __expf(x.y - mx);
      x.z = __expf(x.z - mx); x.w = __expf(x.w - mx);
      *(float4*)(srow + i) = x;
      prow[(i >> 1) + 0] = pack2(x.x, x.y);
      prow[(i >> 1) + 1] = pack2(x.z, x.w);
      sum += x.x + x.y + x.z + x.w;
    }
    sRed[t] = sum;
  }
  __syncthreads();
  if (t < 16) {
    float s = 0.f;
    for (int j = 0; j < 16; ++j) s += sRed[t * 16 + j];
    sL[t] = 1.0f / s;
  }
  __syncthreads();

  // ---- pass 3: normalize P in LDS, then async-copy LDS -> global ----------
  {
    const float inv = sL[m];
    for (int i = 0; i < 128; i += 4) {          // scale in place
      float4 x = *(const float4*)(srow + i);
      x.x *= inv; x.y *= inv; x.z *= inv; x.w *= inv;
      *(float4*)(srow + i) = x;
    }
    wait_dscnt0();                              // LDS writes visible to async engine
    unsigned long long gaddr =
        (unsigned long long)(aln + ((size_t)b * TQ_ + q0 + m) * TK_ + seg);
    unsigned int laddr = (unsigned int)(uintptr_t)srow;  // low 32 bits = LDS offset
    for (int i = 0; i < 128; i += 4) {
      asm volatile("global_store_async_from_lds_b128 %0, %1, off"
                   :: "v"(gaddr + (unsigned long long)i * 4u),
                      "v"(laddr + (unsigned int)i * 4u)
                   : "memory");
    }
    wait_asynccnt0();                           // region aliased by sVt below
  }

  // ---- Phase B: context = P @ V (wave owns d-slice [128*wave, +128)) ------
  v8f co[8] = {};
  const int kp2  = t & 15;          // k-pair slot 0..15
  const int dseg = (t >> 4) * 64;   // d range for cooperative V transpose
  const uint4* pa_row = (const uint4*)(sP + (size_t)lq * SP_STRIDE);

  for (int kc = 0; kc < TK_; kc += 32) {
    __syncthreads();                // all waves past async copy / prior reads
    { // cooperative load of V[kc..kc+32) x D, transposed to bf16 pairs in LDS
      const float* v0 = Eb + (size_t)(kc + 2 * kp2) * D_ + dseg;
      const float* v1 = v0 + D_;
      unsigned int* dst = sVt + kp2;
      for (int i = 0; i < 64; i += 4) {
        float4 x0 = *(const float4*)(v0 + i);
        float4 x1 = *(const float4*)(v1 + i);
        dst[(size_t)(dseg + i + 0) * SVT_STRIDE_U] = pack2(x0.x, x1.x);
        dst[(size_t)(dseg + i + 1) * SVT_STRIDE_U] = pack2(x0.y, x1.y);
        dst[(size_t)(dseg + i + 2) * SVT_STRIDE_U] = pack2(x0.z, x1.z);
        dst[(size_t)(dseg + i + 3) * SVT_STRIDE_U] = pack2(x0.w, x1.w);
      }
    }
    __syncthreads();
    BF16x16 pa;                     // A = P tile (same for all 8 n-tiles)
    pa.q[0] = pa_row[(kc + 16 * hw) >> 3];
    pa.q[1] = pa_row[(kc + 16 * hw + 8) >> 3];
    #pragma unroll
    for (int n = 0; n < 8; ++n) {
      const int d = 128 * wave + 16 * n + lq;   // B col = output d
      BF16x16 vb;
      const uint4* vrow = (const uint4*)(sVt + (size_t)d * SVT_STRIDE_U);
      vb.q[0] = vrow[hw * 2 + 0];
      vb.q[1] = vrow[hw * 2 + 1];
      co[n] = __builtin_amdgcn_wmma_f32_16x16x32_bf16(false, pa.v, false, vb.v,
                                                      (short)0, co[n], false, false);
    }
  }

  // ---- Store context, folding in 1/l --------------------------------------
  #pragma unroll
  for (int r = 0; r < 8; ++r) {
    const float inv = sL[r + 8 * hw];
    float* crow = ctx + ((size_t)b * TQ_ + q0 + r + 8 * hw) * D_ + 128 * wave + lq;
    #pragma unroll
    for (int n = 0; n < 8; ++n)
      crow[16 * n] = co[n][r] * inv;
  }
}

extern "C" void kernel_launch(void* const* d_in, const int* in_sizes, int n_in,
                              void* d_out, int out_size, void* d_ws, size_t ws_size,
                              hipStream_t stream) {
  const float* dec = (const float*)d_in[0];   // [B, TQ, D] f32
  const float* enc = (const float*)d_in[1];   // [B, TK, D] f32
  float* ctx = (float*)d_out;                               // [B, TQ, D]
  float* aln = ctx + (size_t)B_ * TQ_ * D_;                 // [B, TQ, TK]

  (void)hipFuncSetAttribute(
      reinterpret_cast<const void*>(LuongAttention_17815524344562_kernel),
      hipFuncAttributeMaxDynamicSharedMemorySize, SMEM_BYTES);

  dim3 grid(TQ_ / 16, B_);
  LuongAttention_17815524344562_kernel<<<grid, 256, SMEM_BYTES, stream>>>(
      dec, enc, ctx, aln);
}
